// LigandGINE_59554016526995
// MI455X (gfx1250) — matrix operand order repaired and checked by reading, not compiled, not run on hardware
//
#include <hip/hip_runtime.h>
#include <hip/hip_bf16.h>

#define H      256
#define LCOUNT 4
#define FEATN  10
#define AS     264   // LDS row stride in bf16 elements (256 + 8 pad)

typedef __attribute__((ext_vector_type(16))) __bf16 v16bf;
typedef __attribute__((ext_vector_type(8)))  float  v8f;
typedef __attribute__((ext_vector_type(4)))  float  f4x;   // native vec for NT builtins

union BF16Frag { v16bf v; unsigned u[8]; };

// silu via hardware v_rcp_f32 instead of IEEE divide (v_div_scale chains)
__device__ __forceinline__ float silu_(float x) {
  return x * __builtin_amdgcn_rcpf(1.f + __expf(-x));
}

__device__ __forceinline__ float wsum_(float v) {
#pragma unroll
  for (int m = 16; m > 0; m >>= 1) v += __shfl_xor(v, m, 32);
  return v;
}

// ---------------------------------------------------------------------------
// x = LayerNorm(silu(emb[z] + feat @ Wf + bf), g0, b0)   one wave per node
// ---------------------------------------------------------------------------
__global__ void __launch_bounds__(256)
node_init(const int* __restrict__ z, const float* __restrict__ feat,
          const float* __restrict__ emb, const float* __restrict__ Wf,
          const float* __restrict__ bfv, const float* __restrict__ g0,
          const float* __restrict__ b0, float* __restrict__ x, int N)
{
  const int wave = threadIdx.x >> 5, lane = threadIdx.x & 31;
  const int n = blockIdx.x * 8 + wave;
  if (n >= N) return;
  const int zz = z[n];
  float f[FEATN];
#pragma unroll
  for (int t = 0; t < FEATN; ++t) f[t] = feat[n * FEATN + t];
  float a[8];
#pragma unroll
  for (int i = 0; i < 8; ++i) {
    const int h = lane + 32 * i;
    float acc = emb[zz * H + h] + bfv[h];
#pragma unroll
    for (int t = 0; t < FEATN; ++t) acc = fmaf(f[t], Wf[t * H + h], acc);
    a[i] = silu_(acc);
  }
  float s = 0.f, s2 = 0.f;
#pragma unroll
  for (int i = 0; i < 8; ++i) { s += a[i]; s2 += a[i] * a[i]; }
  s = wsum_(s); s2 = wsum_(s2);
  const float mu  = s * (1.f / (float)H);
  const float var = s2 * (1.f / (float)H) - mu * mu;
  const float inv = rsqrtf(var + 1e-5f);
#pragma unroll
  for (int i = 0; i < 8; ++i) {
    const int h = lane + 32 * i;
    x[(long long)n * H + h] = (a[i] - mu) * inv * g0[h] + b0[h];
  }
}

// ---------------------------------------------------------------------------
// e = silu(edge_attr @ We + be)   thread per (edge, 4 channels)
// e is streamed once per layer (410 MB) -> non-temporal store keeps x in L2
// ---------------------------------------------------------------------------
__global__ void __launch_bounds__(256)
edge_proj(const float* __restrict__ ea, const float* __restrict__ We,
          const float* __restrict__ be, float* __restrict__ e, int E)
{
  const long long idx = (long long)blockIdx.x * blockDim.x + threadIdx.x;
  if (idx >= (long long)E * (H / 4)) return;
  const int eid = (int)(idx >> 6);
  const int c   = ((int)idx & 63) << 2;
  const float a0 = ea[eid * 4 + 0], a1 = ea[eid * 4 + 1];
  const float a2 = ea[eid * 4 + 2], a3 = ea[eid * 4 + 3];
  const f4x w0 = *(const f4x*)(We + 0 * H + c);
  const f4x w1 = *(const f4x*)(We + 1 * H + c);
  const f4x w2 = *(const f4x*)(We + 2 * H + c);
  const f4x w3 = *(const f4x*)(We + 3 * H + c);
  const f4x bb = *(const f4x*)(be + c);
  f4x r;
  r.x = silu_(a0 * w0.x + a1 * w1.x + a2 * w2.x + a3 * w3.x + bb.x);
  r.y = silu_(a0 * w0.y + a1 * w1.y + a2 * w2.y + a3 * w3.y + bb.y);
  r.z = silu_(a0 * w0.z + a1 * w1.z + a2 * w2.z + a3 * w3.z + bb.z);
  r.w = silu_(a0 * w0.w + a1 * w1.w + a2 * w2.w + a3 * w3.w + bb.w);
  __builtin_nontemporal_store(r, (f4x*)(e + (long long)eid * H + c));
}

// ---------------------------------------------------------------------------
// agg := x   (GINE eps=0: MLP input is x + segment_sum, so seed with x)
// ---------------------------------------------------------------------------
__global__ void __launch_bounds__(256)
seed_agg(const f4x* __restrict__ x, f4x* __restrict__ agg, long long n4)
{
  const long long i = (long long)blockIdx.x * blockDim.x + threadIdx.x;
  if (i < n4) agg[i] = x[i];
}

// ---------------------------------------------------------------------------
// agg[dst] += relu(x[src] + e)    fp32 global atomics, e read non-temporal
// ---------------------------------------------------------------------------
__global__ void __launch_bounds__(256)
msg_scatter(const int* __restrict__ srcI, const int* __restrict__ dstI,
            const float* __restrict__ x, const float* __restrict__ e,
            float* __restrict__ agg, int E)
{
  const long long idx = (long long)blockIdx.x * blockDim.x + threadIdx.x;
  if (idx >= (long long)E * (H / 4)) return;
  const int eid = (int)(idx >> 6);
  const int c   = ((int)idx & 63) << 2;
  const int s = srcI[eid], d = dstI[eid];
  const f4x xv = *(const f4x*)(x + (long long)s * H + c);
  const f4x ev = __builtin_nontemporal_load((const f4x*)(e + (long long)eid * H + c));
  f4x m;
  m.x = fmaxf(xv.x + ev.x, 0.f);
  m.y = fmaxf(xv.y + ev.y, 0.f);
  m.z = fmaxf(xv.z + ev.z, 0.f);
  m.w = fmaxf(xv.w + ev.w, 0.f);
  float* p = agg + (long long)d * H + c;
  unsafeAtomicAdd(p + 0, m.x);
  unsafeAtomicAdd(p + 1, m.y);
  unsafeAtomicAdd(p + 2, m.z);
  unsafeAtomicAdd(p + 3, m.w);
}

// ---------------------------------------------------------------------------
// W[l][k][n] (f32) -> Wb[l][n][k] (bf16), K contiguous for u32 pair loads
// ---------------------------------------------------------------------------
__global__ void __launch_bounds__(256)
wconv(const float* __restrict__ W, __bf16* __restrict__ Wb, int total)
{
  const int idx = blockIdx.x * blockDim.x + threadIdx.x;
  if (idx >= total) return;
  const int l = idx >> 16;        // / (H*H)
  const int r = idx & 65535;
  const int k = r >> 8, n = r & 255;
  Wb[(l << 16) + (n << 8) + k] = (__bf16)W[idx];
}

// ---------------------------------------------------------------------------
// Fused MLP: hbuf = silu(agg @ W1 + b1) @ W2 + b2   (agg already = x + sum)
// 8 waves/block, 16 rows/wave, wave-private LDS tiles, bf16 WMMA.
// ---------------------------------------------------------------------------
__global__ void __launch_bounds__(256)
gine_mlp(const float* __restrict__ agg,
         const __bf16* __restrict__ W1b, const float* __restrict__ b1,
         const __bf16* __restrict__ W2b, const float* __restrict__ b2,
         float* __restrict__ hbuf, int N)
{
  extern __shared__ __bf16 lds[];
  const int wave = threadIdx.x >> 5, lane = threadIdx.x & 31;
  const int l15 = lane & 15;   // A row index / C,D column index
  const int kh  = lane >> 4;   // K-half selector for A/B fragments
  const int row0 = blockIdx.x * 128 + wave * 16;

  // Pre-warm L2 with this layer's weights (global_prefetch_b8):
  // 256 threads x 256 elements == full 64K-element bf16 matrix each.
  __builtin_prefetch(W1b + threadIdx.x * 256, 0, 1);
  __builtin_prefetch(W2b + threadIdx.x * 256, 0, 1);

  __bf16* Asb = lds + wave * 16 * AS;               // input tile 16 x 256
  __bf16* Tsb = lds + 8 * 16 * AS + wave * 16 * AS; // mid tile   16 x 256

  // Stage A = bf16(agg), coalesced within the wave.
  for (int t = lane; t < 16 * H; t += 32) {
    const int r = t >> 8, c = t & (H - 1);
    const int gr = row0 + r;
    float v = (gr < N) ? agg[(long long)gr * H + c] : 0.f;
    Asb[r * AS + c] = (__bf16)v;
  }
  // No barrier needed: tiles are wave-private; DS ops are in-order per wave.

  // ---- GEMM1: T = silu(A @ W1 + b1), restaged to LDS as bf16 ----
  for (int j = 0; j < 16; ++j) {
    const float bias = b1[j * 16 + l15];
    v8f acc;
#pragma unroll
    for (int r = 0; r < 8; ++r) acc[r] = bias;
#pragma unroll
    for (int kk = 0; kk < 8; ++kk) {
      BF16Frag a, b;
#pragma unroll
      for (int q = 0; q < 8; ++q) {
        const int ka = ((q & 4) ? 16 : 0) + kh * 8 + ((q & 3) << 1);
        a.u[q] = *(const unsigned*)(Asb + l15 * AS + kk * 32 + ka);
        b.u[q] = *(const unsigned*)(W1b + (j * 16 + l15) * H + kk * 32 + ka);
      }
      acc = __builtin_amdgcn_wmma_f32_16x16x32_bf16(false, a.v, false, b.v,
                                                    (short)0, acc, false, false);
    }
#pragma unroll
    for (int r = 0; r < 8; ++r)
      Tsb[(kh * 8 + r) * AS + j * 16 + l15] = (__bf16)silu_(acc[r]);
  }

  // ---- GEMM2: h = T @ W2 + b2 ----
  for (int j = 0; j < 16; ++j) {
    const float bias = b2[j * 16 + l15];
    v8f acc;
#pragma unroll
    for (int r = 0; r < 8; ++r) acc[r] = bias;
#pragma unroll
    for (int kk = 0; kk < 8; ++kk) {
      BF16Frag a, b;
#pragma unroll
      for (int q = 0; q < 8; ++q) {
        const int ka = ((q & 4) ? 16 : 0) + kh * 8 + ((q & 3) << 1);
        a.u[q] = *(const unsigned*)(Tsb + l15 * AS + kk * 32 + ka);
        b.u[q] = *(const unsigned*)(W2b + (j * 16 + l15) * H + kk * 32 + ka);
      }
      acc = __builtin_amdgcn_wmma_f32_16x16x32_bf16(false, a.v, false, b.v,
                                                    (short)0, acc, false, false);
    }
#pragma unroll
    for (int r = 0; r < 8; ++r) {
      const int gr = row0 + kh * 8 + r;
      if (gr < N) hbuf[(long long)gr * H + j * 16 + l15] = acc[r];
    }
  }
}

// ---------------------------------------------------------------------------
// x = LayerNorm(x + hbuf, g, b)   one wave per node
// ---------------------------------------------------------------------------
__global__ void __launch_bounds__(256)
ln_apply(float* __restrict__ x, const float* __restrict__ hbuf,
         const float* __restrict__ g, const float* __restrict__ b, int N)
{
  const int wave = threadIdx.x >> 5, lane = threadIdx.x & 31;
  const int n = blockIdx.x * 8 + wave;
  if (n >= N) return;
  float a[8];
#pragma unroll
  for (int i = 0; i < 8; ++i) {
    const int h = lane + 32 * i;
    a[i] = x[(long long)n * H + h] + hbuf[(long long)n * H + h];
  }
  float s = 0.f, s2 = 0.f;
#pragma unroll
  for (int i = 0; i < 8; ++i) { s += a[i]; s2 += a[i] * a[i]; }
  s = wsum_(s); s2 = wsum_(s2);
  const float mu  = s * (1.f / (float)H);
  const float var = s2 * (1.f / (float)H) - mu * mu;
  const float inv = rsqrtf(var + 1e-5f);
#pragma unroll
  for (int i = 0; i < 8; ++i) {
    const int h = lane + 32 * i;
    x[(long long)n * H + h] = (a[i] - mu) * inv * g[h] + b[h];
  }
}

// ---------------------------------------------------------------------------
__global__ void __launch_bounds__(256)
copy_batch(const int* __restrict__ batch, int* __restrict__ out, int N)
{
  const int i = blockIdx.x * blockDim.x + threadIdx.x;
  if (i < N) out[i] = batch[i];
}

// ===========================================================================
extern "C" void kernel_launch(void* const* d_in, const int* in_sizes, int n_in,
                              void* d_out, int out_size, void* d_ws, size_t ws_size,
                              hipStream_t stream)
{
  const int*   z     = (const int*)  d_in[0];
  const float* feat  = (const float*)d_in[1];
  const int*   ei    = (const int*)  d_in[2];
  const float* ea    = (const float*)d_in[3];
  const int*   batch = (const int*)  d_in[4];
  const float* emb   = (const float*)d_in[5];
  const float* Wf    = (const float*)d_in[6];
  const float* bfv   = (const float*)d_in[7];
  const float* g0    = (const float*)d_in[8];
  const float* b0    = (const float*)d_in[9];
  const float* We    = (const float*)d_in[10];
  const float* be    = (const float*)d_in[11];
  const float* W1    = (const float*)d_in[12];
  const float* b1    = (const float*)d_in[13];
  const float* W2    = (const float*)d_in[14];
  const float* b2    = (const float*)d_in[15];
  const float* lng   = (const float*)d_in[16];
  const float* lnb   = (const float*)d_in[17];

  const int N = in_sizes[0];
  const int E = in_sizes[2] / 2;
  const int* srcI = ei;
  const int* dstI = ei + E;

  float* x    = (float*)d_out;                 // [N,H] output lives here
  int*   bout = (int*)(x + (size_t)N * H);     // batch appended after x

  float*  agg  = (float*)d_ws;
  float*  hbuf = agg  + (size_t)N * H;
  float*  ebuf = hbuf + (size_t)N * H;
  __bf16* W1b  = (__bf16*)(ebuf + (size_t)E * H);
  __bf16* W2b  = W1b + (size_t)LCOUNT * H * H;

  const dim3 blk(256);

  { // weight conversion to bf16, transposed [l][n][k]
    const int tot = LCOUNT * H * H;
    const int g = (tot + 255) / 256;
    wconv<<<g, blk, 0, stream>>>(W1, W1b, tot);
    wconv<<<g, blk, 0, stream>>>(W2, W2b, tot);
  }

  node_init<<<(N + 7) / 8, blk, 0, stream>>>(z, feat, emb, Wf, bfv, g0, b0, x, N);

  {
    const long long tot = (long long)E * (H / 4);
    const int g = (int)((tot + 255) / 256);
    edge_proj<<<g, blk, 0, stream>>>(ea, We, be, ebuf, E);
  }

  const size_t shmem = (size_t)2 * 8 * 16 * AS * sizeof(__bf16); // 135168 B

  for (int l = 0; l < LCOUNT; ++l) {
    {
      const long long n4 = (long long)N * H / 4;
      const int g = (int)((n4 + 255) / 256);
      seed_agg<<<g, blk, 0, stream>>>((const f4x*)x, (f4x*)agg, n4);
    }
    {
      const long long tot = (long long)E * (H / 4);
      const int g = (int)((tot + 255) / 256);
      msg_scatter<<<g, blk, 0, stream>>>(srcI, dstI, x, ebuf, agg, E);
    }
    {
      const int g = (N + 127) / 128;
      gine_mlp<<<g, blk, shmem, stream>>>(agg,
                                          W1b + (size_t)l * H * H, b1 + l * H,
                                          W2b + (size_t)l * H * H, b2 + l * H,
                                          hbuf, N);
    }
    ln_apply<<<(N + 7) / 8, blk, 0, stream>>>(x, hbuf, lng + l * H, lnb + l * H, N);
  }

  copy_batch<<<(N + 255) / 256, blk, 0, stream>>>(batch, bout, N);
}
